// EntityCell_8770323219172
// MI455X (gfx1250) — matrix operand order repaired and checked by reading, not compiled, not run on hardware
//
#include <hip/hip_runtime.h>

// CDNA5 / gfx1250: wave32, WMMA 16x16x32 bf16 -> f32 accumulate.
typedef __attribute__((ext_vector_type(16))) __bf16 v16bf;
typedef __attribute__((ext_vector_type(8)))  float  v8f;

#define B_N 8192
#define E_N 20
#define D_N 128
#define ROWS 16
#define NTILES ((B_N * E_N) / ROWS)   // 10240 row-tiles of 16 (b,e) rows
#define GRID_MAIN 2560                // each block walks 4 tiles (grid-stride)

union Frag {
  uint4 q[2];          // 8 dwords = 16 bf16
  unsigned int u[8];
  v16bf v;
};

__device__ __forceinline__ unsigned pack_bf16(float a, float b) {
  // round-to-nearest-even f32 -> bf16, packed pair (lo, hi)
  unsigned ua = __float_as_uint(a);
  unsigned ub = __float_as_uint(b);
  ua = (ua + 0x7FFFu + ((ua >> 16) & 1u)) >> 16;
  ub = (ub + 0x7FFFu + ((ub >> 16) & 1u)) >> 16;
  return ua | (ub << 16);
}

// Branch-free tanh using hardware transcendentals (v_exp_f32 + v_rcp_f32).
// tanh(x) = 1 - 2/(exp(2x)+1); saturates correctly at +/-inf.
__device__ __forceinline__ float fast_tanh(float x) {
  float e = __expf(2.f * x);
  return 1.f - 2.f * __builtin_amdgcn_rcpf(e + 1.f);
}

// Branch-free sigmoid: v_exp_f32 + v_rcp_f32, no div_fixup chain.
__device__ __forceinline__ float fast_sigmoid(float x) {
  return __builtin_amdgcn_rcpf(1.f + __expf(-x));
}

// ---------------------------------------------------------------------------
// Prep: convert U, V, W (f32 128x128) into bf16 B-matrix fragment layout.
// Fragment layout per matrix (8192 dwords = 32KB):
//   idx = ((k0*8 + n0)*32 + lane)*8 + v
//   lane 0-15  -> N = n0*16 + lane,      K = k0*32 + v*2 + {0,1}
//   lane 16-31 -> N = n0*16 + (lane-16), K = k0*32 + 16 + v*2 + {0,1}
// so each wave's 32x16 B tile is two contiguous uint4 loads per lane.
// ---------------------------------------------------------------------------
__global__ void prep_weight_frags(const float* __restrict__ U,
                                  const float* __restrict__ V,
                                  const float* __restrict__ W,
                                  unsigned int* __restrict__ frag) {
  int idx  = blockIdx.x * blockDim.x + threadIdx.x;  // 0 .. 24575
  int mat  = idx >> 13;                              // 0=U 1=V 2=W
  int r    = idx & 8191;
  int v    = r & 7;
  int lane = (r >> 3) & 31;
  int n0   = (r >> 8) & 7;
  int k0   = r >> 11;
  int n = n0 * 16 + (lane & 15);
  int k = k0 * 32 + (lane >> 4) * 16 + v * 2;
  const float* X = (mat == 0) ? U : (mat == 1) ? V : W;
  frag[idx] = pack_bf16(X[k * D_N + n], X[(k + 1) * D_N + n]);
}

// ---------------------------------------------------------------------------
// Main fused kernel: one block = 8 waves, one row-tile of 16 (b,e) rows.
// Wave w computes columns [16w, 16w+16) of D via 12 WMMAs/tile.
// A-fragment ds_loads are software-pipelined one k-step ahead so the WMMAs
// overlap LDS latency (partial s_wait_dscnt instead of dscnt==0 each step).
// ---------------------------------------------------------------------------
__global__ __launch_bounds__(256)
void entity_cell_main(const float* __restrict__ enc,
                      const float* __restrict__ prev,
                      const float* __restrict__ keys,
                      const uint4* __restrict__ wfrag,
                      float* __restrict__ out) {
  __shared__ __align__(16) float sPrev[ROWS * D_N];   // 8 KB
  __shared__ __align__(16) float sKeys[ROWS * D_N];   // 8 KB
  __shared__ __align__(16) float sEnc [ROWS * D_N];   // 8 KB
  __shared__ __align__(16) unsigned int sAp[1024];    // bf16 A-frags (prev)
  __shared__ __align__(16) unsigned int sAk[1024];    // bf16 A-frags (keys)
  __shared__ __align__(16) unsigned int sAe[1024];    // bf16 A-frags (enc)
  __shared__ float sPart[256];
  __shared__ float sGates[ROWS];
  __shared__ float sRowSum[ROWS];

  const int tid  = threadIdx.x;
  const int lane = tid & 31;
  const int w    = tid >> 5;   // wave id 0..7
  const int n0   = w * 16;     // this wave's D-column tile

  const uint4* Uf = wfrag;          // 2048 uint4 each
  const uint4* Vf = wfrag + 2048;
  const uint4* Wf = wfrag + 4096;

  for (int t = blockIdx.x; t < NTILES; t += GRID_MAIN) {
    const int r0 = t * ROWS;

    // ---- stage 16 rows of prev/keys/enc into LDS (coalesced float4) ----
    for (int i = tid; i < ROWS * D_N / 4; i += 256) {
      int row = i >> 5;
      int c4  = i & 31;
      size_t r = (size_t)(r0 + row);
      size_t b = r / E_N;
      ((float4*)sPrev)[i] = ((const float4*)(prev + r * D_N))[c4];
      ((float4*)sKeys)[i] = ((const float4*)(keys + r * D_N))[c4];
      ((float4*)sEnc )[i] = ((const float4*)(enc  + b * D_N))[c4];
    }
    // ---- prefetch next tile's rows into cache (global_prefetch_b8) ----
    int tn = t + GRID_MAIN;
    if (tn < NTILES) {
      size_t rn0 = (size_t)tn * ROWS;
      if (tid < 16)      __builtin_prefetch(prev + (rn0 + tid) * D_N, 0, 3);
      else if (tid < 32) __builtin_prefetch(keys + (rn0 + (tid - 16)) * D_N, 0, 3);
      else if (tid < 48) __builtin_prefetch(enc + ((rn0 + (tid - 32)) / E_N) * D_N, 0, 3);
    }
    __syncthreads();

    // ---- gate partials: gate[r] = sigmoid(sum_d enc*(prev+keys)) ----
    {
      int grow = tid >> 4, gseg = tid & 15;
      const float* ep = &sEnc [grow * D_N + gseg * 8];
      const float* pp = &sPrev[grow * D_N + gseg * 8];
      const float* kp = &sKeys[grow * D_N + gseg * 8];
      float s = 0.f;
#pragma unroll
      for (int j = 0; j < 8; ++j) s += ep[j] * (pp[j] + kp[j]);
      sPart[tid] = s;
    }
    // ---- convert rows to bf16 A-fragments in LDS (contiguous k = 16*half + 2v order, same as B frags) ----
    for (int i = tid; i < 1024; i += 256) {
      int v  = i & 7;
      int ln = (i >> 3) & 31;
      int k0 = i >> 8;
      int m  = ln & 15;
      int kh = ln >> 4;  // lane half selects K sub-stripe
      int kl = kh * 16 + v * 2;
      int a  = m * D_N + k0 * 32 + kl;
      sAp[i] = pack_bf16(sPrev[a], sPrev[a + 1]);
      sAk[i] = pack_bf16(sKeys[a], sKeys[a + 1]);
      sAe[i] = pack_bf16(sEnc [a], sEnc [a + 1]);
    }
    __syncthreads();
    if (tid < 16) {
      float g = 0.f;
#pragma unroll
      for (int j = 0; j < 16; ++j) g += sPart[tid * 16 + j];
      sGates[tid]  = fast_sigmoid(g);
      sRowSum[tid] = 0.f;
    }
    __syncthreads();

    // ---- C = prev@U + keys@V + enc@W via v_wmma_f32_16x16x32_bf16 ----
    // Software-pipelined: A-frags for step k+1 are loaded while step k's
    // WMMAs execute. B-frags live in registers across the whole tile loop.
    v8f acc = {0.f, 0.f, 0.f, 0.f, 0.f, 0.f, 0.f, 0.f};
    const uint4* ap4 = (const uint4*)sAp;
    const uint4* ak4 = (const uint4*)sAk;
    const uint4* ae4 = (const uint4*)sAe;

    Frag aP, aK, aE;
    {
      int ao = lane * 2;                                 // k0 = 0
      aP.q[0] = ap4[ao]; aP.q[1] = ap4[ao + 1];
      aK.q[0] = ak4[ao]; aK.q[1] = ak4[ao + 1];
      aE.q[0] = ae4[ao]; aE.q[1] = ae4[ao + 1];
    }
#pragma unroll
    for (int k0 = 0; k0 < 4; ++k0) {
      Frag nP, nK, nE;
      if (k0 < 3) {                                      // prefetch k0+1
        int an = ((k0 + 1) * 32 + lane) * 2;
        nP.q[0] = ap4[an]; nP.q[1] = ap4[an + 1];
        nK.q[0] = ak4[an]; nK.q[1] = ak4[an + 1];
        nE.q[0] = ae4[an]; nE.q[1] = ae4[an + 1];
      }
      Frag bU, bV, bW;
      int bo = ((k0 * 8 + w) * 32 + lane) * 2;           // hoisted by compiler
      bU.q[0] = Uf[bo]; bU.q[1] = Uf[bo + 1];
      bV.q[0] = Vf[bo]; bV.q[1] = Vf[bo + 1];
      bW.q[0] = Wf[bo]; bW.q[1] = Wf[bo + 1];
      acc = __builtin_amdgcn_wmma_f32_16x16x32_bf16(false, aP.v, false, bU.v,
                                                    (short)0, acc, false, false);
      acc = __builtin_amdgcn_wmma_f32_16x16x32_bf16(false, aK.v, false, bV.v,
                                                    (short)0, acc, false, false);
      acc = __builtin_amdgcn_wmma_f32_16x16x32_bf16(false, aE.v, false, bW.v,
                                                    (short)0, acc, false, false);
      if (k0 < 3) { aP = nP; aK = nK; aE = nE; }         // renamed away (unrolled)
    }

    // ---- epilogue: tanh, gated residual, row-norm reduction ----
    // C layout: vgpr j, lanes 0-15 -> (M=j, N=lane); lanes 16-31 -> (M=j+8)
    float up[8];
#pragma unroll
    for (int j = 0; j < 8; ++j) {
      int m   = (lane < 16) ? j : (j + 8);
      float h = fast_tanh(acc[j]);
      float u = sPrev[m * D_N + n0 + (lane & 15)] + sGates[m] * h;
      up[j]   = u;
      float sq = u * u;
#pragma unroll
      for (int off = 1; off < 16; off <<= 1)      // reduce within each 16-lane half
        sq += __shfl_xor(sq, off, 32);
      if ((lane & 15) == 0) atomicAdd(&sRowSum[m], sq);  // ds_add_f32
    }
    __syncthreads();

#pragma unroll
    for (int j = 0; j < 8; ++j) {
      int m = (lane < 16) ? j : (j + 8);
      float inv = __builtin_amdgcn_rsqf(fmaxf(sRowSum[m], 1e-12f));  // v_rsq_f32
      out[(size_t)(r0 + m) * D_N + n0 + (lane & 15)] = up[j] * inv;
    }
    __syncthreads();   // protect LDS before next tile
  }
}

// ---------------------------------------------------------------------------
extern "C" void kernel_launch(void* const* d_in, const int* in_sizes, int n_in,
                              void* d_out, int out_size, void* d_ws, size_t ws_size,
                              hipStream_t stream) {
  const float* enc  = (const float*)d_in[0];   // (B, D)
  const float* prev = (const float*)d_in[1];   // (B, E, D)
  const float* keys = (const float*)d_in[2];   // (B, E, D)
  const float* U    = (const float*)d_in[3];   // (D, D)
  const float* V    = (const float*)d_in[4];   // (D, D)
  const float* W    = (const float*)d_in[5];   // (D, D)
  float* out = (float*)d_out;                  // (B, E, D)
  unsigned int* frag = (unsigned int*)d_ws;    // 96 KB bf16 weight fragments

  prep_weight_frags<<<96, 256, 0, stream>>>(U, V, W, frag);
  entity_cell_main<<<GRID_MAIN, 256, 0, stream>>>(enc, prev, keys,
                                                  (const uint4*)frag, out);
}
